// GKDFeatureExtractor_20169166422638
// MI455X (gfx1250) — matrix operand order repaired and checked by reading, not compiled, not run on hardware
//
#include <hip/hip_runtime.h>
#include <hip/hip_bf16.h>

// Problem sizes (fixed by the reference)
#define NW 8192
#define NT 4096
#define HD 128

typedef __attribute__((ext_vector_type(16))) __bf16 v16bf;
typedef __attribute__((ext_vector_type(8)))  float  v8f;
typedef __attribute__((ext_vector_type(4)))  float  v4f;

// ---------------- helpers ----------------

__device__ __forceinline__ __bf16 f2bf(float f) { return (__bf16)f; }

// CDNA5 WMMA bf16: D = A(16x32) * B(32x16) + C, fp32 accum
__device__ __forceinline__ v8f wmma_bf16(v16bf a, v16bf b, v8f c) {
  return __builtin_amdgcn_wmma_f32_16x16x32_bf16(
      /*neg_a=*/false, a, /*neg_b=*/false, b,
      /*c_mod=*/(short)0, c, /*reuse_a=*/false, /*reuse_b=*/false);
}

// Packed B layout for a [K x HD] bf16 matrix: tiles of 32(K) x 16(N), 512 elems.
// Within a tile, lane l owns n = l&15; lane half = (k>>3)&1, element e = 8*((k>>4)&1)+(k&7).
// A consumer B-fragment load is then 32 contiguous bytes per lane.
__device__ __host__ __forceinline__ long bidx(int k, int c) {
  long tile = (long)((k >> 5) * (HD >> 4) + (c >> 4));
  int lane = (c & 15) + (((k >> 3) & 1) << 4);
  return (tile << 9) + ((long)lane << 4) + (((k >> 4) & 1) << 3) + (k & 7);
}

__device__ __forceinline__ v16bf load_b_packed(const __bf16* B, int k0, int col0) {
  const int lane = threadIdx.x & 31;
  const __bf16* p = B + (((long)((k0 >> 5) * (HD >> 4) + (col0 >> 4))) << 9) + ((long)lane << 4);
  return *(const v16bf*)p;
}

// Packed A layout for an [M x Kdim] bf16 matrix: tiles of 16(M) x 32(K), 512 elems,
// laid out so a 16x32 A-fragment is 32 contiguous bytes per lane.
// Inverse of the ISA 16-bit A map: lane=(m&15)+16*((k>>3)&1), e=8*((k>>4)&1)+(k&7).
__device__ __forceinline__ long aidx(int m, int k, int Kdim) {
  long tile = (long)((m >> 4) * (Kdim >> 5) + (k >> 5));
  int lane = (m & 15) + (((k >> 3) & 1) << 4);
  int e = (((k >> 4) & 1) << 3) + (k & 7);
  return (tile << 9) + ((long)lane << 4) + e;
}

__device__ __forceinline__ v16bf load_a_packed(const __bf16* A, int Kdim, int row0, int k0) {
  const int lane = threadIdx.x & 31;
  const __bf16* p = A + (((long)((row0 >> 4) * (Kdim >> 5) + (k0 >> 5))) << 9) + ((long)lane << 4);
  return *(const v16bf*)p;
}

// A-fragment (16x32) from row-major fp32. ISA layout:
// lanes 0-15: M=lane, elems 0..7 -> K 0..7,  elems 8..15 -> K 16..23
// lanes16-31: M=lane-16, elems 0..7 -> K 8..15, elems 8..15 -> K 24..31
// => per lane: two contiguous 8-float runs at k0+8g and k0+16+8g  (4x b128 loads)
template <bool NTLOAD>
__device__ __forceinline__ v16bf load_a_f32(const float* A, long lda, int row0, int k0) {
  const int lane = threadIdx.x & 31;
  const int m = lane & 15;
  const int g = lane >> 4;
  const float* p = A + (long)(row0 + m) * lda + k0 + (g << 3);
  v4f f0, f1, f2, f3;
  if (NTLOAD) {
    f0 = __builtin_nontemporal_load((const v4f*)(p + 0));
    f1 = __builtin_nontemporal_load((const v4f*)(p + 4));
    f2 = __builtin_nontemporal_load((const v4f*)(p + 16));
    f3 = __builtin_nontemporal_load((const v4f*)(p + 20));
  } else {
    f0 = *(const v4f*)(p + 0);
    f1 = *(const v4f*)(p + 4);
    f2 = *(const v4f*)(p + 16);
    f3 = *(const v4f*)(p + 20);
  }
  v16bf a;
#pragma unroll
  for (int i = 0; i < 4; ++i) {
    a[i]      = f2bf(f0[i]);
    a[4 + i]  = f2bf(f1[i]);
    a[8 + i]  = f2bf(f2[i]);
    a[12 + i] = f2bf(f3[i]);
  }
  return a;
}

// 16x16 fp32 C/D tile load per ISA layout: n=lane%16, m=8*(lane/16)+r
__device__ __forceinline__ v8f load_tile_f32(const float* C, long ldc, int row0, int col0) {
  const int lane = threadIdx.x & 31;
  const int n = lane & 15;
  const int mb = (lane >> 4) << 3;
  v8f acc;
#pragma unroll
  for (int r = 0; r < 8; ++r)
    acc[r] = C[(long)(row0 + mb + r) * ldc + col0 + n];
  return acc;
}

// ---------------- kernels ----------------

// Pre-pack a 128x128 f32 weight into packed bf16 B layout
__global__ __launch_bounds__(256) void pack_b_kernel(const float* __restrict__ W,
                                                     __bf16* __restrict__ out) {
  int idx = blockIdx.x * 256 + threadIdx.x;  // 0 .. 16383
  int k = idx >> 7, c = idx & 127;
  out[bidx(k, c)] = f2bf(W[idx]);
}

// Transpose-pack wt_adj [NW x NT] f32 into wt_adj^T as packed-A bf16 [NT x NW].
// Source reads are fully coalesced; binary adjacency is exact in bf16.
__global__ __launch_bounds__(256) void transpose_pack_kernel(
    const float* __restrict__ src, __bf16* __restrict__ dst) {
  long idx = (long)blockIdx.x * 256 + threadIdx.x;   // over NW*NT, t fastest
  int w = (int)(idx >> 12);          // NT = 4096
  int t = (int)(idx & (NT - 1));
  float v = __builtin_nontemporal_load(src + idx);
  dst[aidx(t, w, NW)] = f2bf(v);
}

// h = relu(X @ W + b) ; X:[M,128] f32, W packed bf16 -> f32 out (row-major)
__global__ __launch_bounds__(256) void proj_relu_kernel(
    const float* __restrict__ X, const __bf16* __restrict__ Wp,
    const float* __restrict__ bias, float* __restrict__ outF) {
  const int row0 = (blockIdx.y * blockDim.y + threadIdx.y) * 16;
  const int col0 = blockIdx.x * 16;
  v8f acc = {};
#pragma unroll
  for (int k0 = 0; k0 < HD; k0 += 32)
    acc = wmma_bf16(load_a_f32<false>(X, HD, row0, k0), load_b_packed(Wp, k0, col0), acc);
  const int lane = threadIdx.x & 31;
  const int n = lane & 15;
  const int mb = (lane >> 4) << 3;
  const float bv = bias[col0 + n];
#pragma unroll
  for (int r = 0; r < 8; ++r) {
    float v = acc[r] + bv;
    outF[(long)(row0 + mb + r) * HD + col0 + n] = v > 0.f ? v : 0.f;
  }
}

// C = A @ W (+bias) ; A:[M,128] f32 row-major, W packed bf16.
// Optional outputs: f32 row-major and/or packed-bf16 (B layout for later consumers).
__global__ __launch_bounds__(256) void gemm128_kernel(
    const float* __restrict__ A, const __bf16* __restrict__ Wp,
    const float* __restrict__ bias, float* __restrict__ outF,
    __bf16* __restrict__ outB) {
  const int row0 = (blockIdx.y * blockDim.y + threadIdx.y) * 16;
  const int col0 = blockIdx.x * 16;
  v8f acc = {};
#pragma unroll
  for (int k0 = 0; k0 < HD; k0 += 32)
    acc = wmma_bf16(load_a_f32<false>(A, HD, row0, k0), load_b_packed(Wp, k0, col0), acc);
  const int lane = threadIdx.x & 31;
  const int n = lane & 15;
  const int mb = (lane >> 4) << 3;
  const float bv = bias ? bias[col0 + n] : 0.f;
#pragma unroll
  for (int r = 0; r < 8; ++r) {
    float v = acc[r] + bv;
    int R = row0 + mb + r, C = col0 + n;
    if (outF) outF[(long)R * HD + C] = v;
    if (outB) outB[bidx(R, C)] = f2bf(v);
  }
}

// h_w_rgcn = relu(s_w + ww_adj@m_ww + wt_adj@m_tw) -> f32 row-major
// One wave computes a 16x128 strip: adjacency is streamed exactly once.
__global__ __launch_bounds__(256) void rgcn_worker_kernel(
    const float* __restrict__ s_w, const float* __restrict__ ww_adj,
    const __bf16* __restrict__ m_ww, const float* __restrict__ wt_adj,
    const __bf16* __restrict__ m_tw, float* __restrict__ outF) {
  const int row0 = (blockIdx.x * blockDim.y + threadIdx.y) * 16;
  v8f acc[8];
#pragma unroll
  for (int cb = 0; cb < 8; ++cb) acc[cb] = load_tile_f32(s_w, HD, row0, cb << 4);
  for (int k0 = 0; k0 < NW; k0 += 32) {        // ww_adj @ (h_w @ W_ww)
    v16bf a = load_a_f32<true>(ww_adj, NW, row0, k0);
#pragma unroll
    for (int cb = 0; cb < 8; ++cb)
      acc[cb] = wmma_bf16(a, load_b_packed(m_ww, k0, cb << 4), acc[cb]);
  }
  for (int k0 = 0; k0 < NT; k0 += 32) {        // wt_adj @ (h_t @ W_tw)
    v16bf a = load_a_f32<true>(wt_adj, NT, row0, k0);
#pragma unroll
    for (int cb = 0; cb < 8; ++cb)
      acc[cb] = wmma_bf16(a, load_b_packed(m_tw, k0, cb << 4), acc[cb]);
  }
  const int lane = threadIdx.x & 31;
  const int n = lane & 15;
  const int mb = (lane >> 4) << 3;
#pragma unroll
  for (int cb = 0; cb < 8; ++cb)
#pragma unroll
    for (int r = 0; r < 8; ++r) {
      float v = acc[cb][r];
      outF[(long)(row0 + mb + r) * HD + (cb << 4) + n] = v > 0.f ? v : 0.f;
    }
}

// final_task = relu(s_t + wt_adj^T @ m_wt) + h_t -> f32 into d_out task region.
// wt_adj^T comes pre-transposed in packed-A bf16 layout: contiguous 32B/lane loads.
__global__ __launch_bounds__(256) void rgcn_task_kernel(
    const float* __restrict__ s_t, const __bf16* __restrict__ wtT,
    const __bf16* __restrict__ m_wt, const float* __restrict__ h_t,
    float* __restrict__ out) {
  const int row0 = (blockIdx.x * blockDim.y + threadIdx.y) * 16;
  v8f acc[8];
#pragma unroll
  for (int cb = 0; cb < 8; ++cb) acc[cb] = load_tile_f32(s_t, HD, row0, cb << 4);
  for (int k0 = 0; k0 < NW; k0 += 32) {        // wt_adj.T @ (h_w @ W_wt)
    v16bf a = load_a_packed(wtT, NW, row0, k0);
#pragma unroll
    for (int cb = 0; cb < 8; ++cb)
      acc[cb] = wmma_bf16(a, load_b_packed(m_wt, k0, cb << 4), acc[cb]);
  }
  const int lane = threadIdx.x & 31;
  const int n = lane & 15;
  const int mb = (lane >> 4) << 3;
#pragma unroll
  for (int cb = 0; cb < 8; ++cb)
#pragma unroll
    for (int r = 0; r < 8; ++r) {
      float v = acc[cb][r];
      v = v > 0.f ? v : 0.f;
      long idx = (long)(row0 + mb + r) * HD + (cb << 4) + n;
      out[idx] = v + h_t[idx];
    }
}

// f_src[i] = Wh[i,:]·a_src ; f_dst[i] = Wh[i,:]·a_dst
__global__ __launch_bounds__(256) void fvec_kernel(
    const float* __restrict__ Wh, const float* __restrict__ a_src,
    const float* __restrict__ a_dst, float* __restrict__ f_src,
    float* __restrict__ f_dst, int M) {
  int i = blockIdx.x * blockDim.x + threadIdx.x;
  if (i >= M) return;
  float s = 0.f, d = 0.f;
  const float* w = Wh + (long)i * HD;
#pragma unroll 8
  for (int h = 0; h < HD; ++h) { float v = w[h]; s += v * a_src[h]; d += v * a_dst[h]; }
  f_src[i] = s;
  f_dst[i] = d;
}

// Online-softmax row stats over masked logits e_ij = lrelu(f_src[i]+f_dst[j]) | -9e15
__global__ __launch_bounds__(256) void softmax_stats_kernel(
    const float* __restrict__ adj, const float* __restrict__ f_src,
    const float* __restrict__ f_dst, float* __restrict__ rmax,
    float* __restrict__ rsum) {
  __shared__ float sm[256], ss[256];
  const int i = blockIdx.x;
  const int tid = threadIdx.x;
  const float fs = f_src[i];
  float m = -3.0e38f, s = 0.f;
  for (int j = tid; j < NW; j += 256) {
    float x = fs + f_dst[j];
    x = x > 0.f ? x : 0.2f * x;
    float a = __builtin_nontemporal_load(adj + (long)i * NW + j);
    float logit = a > 0.f ? x : -9.0e15f;
    if (logit > m) { s = s * __expf(m - logit) + 1.f; m = logit; }
    else           { s += __expf(logit - m); }
  }
  sm[tid] = m; ss[tid] = s;
  __syncthreads();
  for (int off = 128; off > 0; off >>= 1) {
    if (tid < off) {
      float m2 = sm[tid + off], s2 = ss[tid + off];
      float M = sm[tid] > m2 ? sm[tid] : m2;
      ss[tid] = ss[tid] * __expf(sm[tid] - M) + s2 * __expf(m2 - M);
      sm[tid] = M;
    }
    __syncthreads();
  }
  if (tid == 0) { rmax[i] = sm[0]; rsum[i] = ss[0]; }
}

// final_worker = elu((P @ Wh)/rowsum) + h_w, P_ij = exp(masked_logit - rowmax),
// built on the fly (flash-attention style). One wave -> 16x128 strip so each
// adjacency element / exp is computed once and feeds 8 WMMAs.
__global__ __launch_bounds__(256) void attn_kernel(
    const float* __restrict__ adj, const __bf16* __restrict__ WhB,
    const float* __restrict__ f_src, const float* __restrict__ f_dst,
    const float* __restrict__ rmax, const float* __restrict__ rsum,
    const float* __restrict__ h_w, float* __restrict__ out) {
  const int row0 = (blockIdx.x * blockDim.y + threadIdx.y) * 16;
  const int lane = threadIdx.x & 31;
  const int mA = lane & 15;
  const int g = lane >> 4;
  const float fs = f_src[row0 + mA];
  const float rm = rmax[row0 + mA];
  const float* arow = adj + (long)(row0 + mA) * NW;
  v8f acc[8];
#pragma unroll
  for (int cb = 0; cb < 8; ++cb) acc[cb] = v8f{};
  for (int k0 = 0; k0 < NW; k0 += 32) {
    const int jb = k0 + (g << 3);
    v4f a0 = __builtin_nontemporal_load((const v4f*)(arow + jb));
    v4f a1 = __builtin_nontemporal_load((const v4f*)(arow + jb + 4));
    v4f a2 = __builtin_nontemporal_load((const v4f*)(arow + jb + 16));
    v4f a3 = __builtin_nontemporal_load((const v4f*)(arow + jb + 20));
    v4f d0 = *(const v4f*)(f_dst + jb);
    v4f d1 = *(const v4f*)(f_dst + jb + 4);
    v4f d2 = *(const v4f*)(f_dst + jb + 16);
    v4f d3 = *(const v4f*)(f_dst + jb + 20);
    v16bf afrag;
#pragma unroll
    for (int i = 0; i < 4; ++i) {
      float x0 = fs + d0[i]; x0 = x0 > 0.f ? x0 : 0.2f * x0;
      float x1 = fs + d1[i]; x1 = x1 > 0.f ? x1 : 0.2f * x1;
      float x2 = fs + d2[i]; x2 = x2 > 0.f ? x2 : 0.2f * x2;
      float x3 = fs + d3[i]; x3 = x3 > 0.f ? x3 : 0.2f * x3;
      afrag[i]      = f2bf(__expf((a0[i] > 0.f ? x0 : -9.0e15f) - rm));
      afrag[4 + i]  = f2bf(__expf((a1[i] > 0.f ? x1 : -9.0e15f) - rm));
      afrag[8 + i]  = f2bf(__expf((a2[i] > 0.f ? x2 : -9.0e15f) - rm));
      afrag[12 + i] = f2bf(__expf((a3[i] > 0.f ? x3 : -9.0e15f) - rm));
    }
#pragma unroll
    for (int cb = 0; cb < 8; ++cb)
      acc[cb] = wmma_bf16(afrag, load_b_packed(WhB, k0, cb << 4), acc[cb]);
  }
  const int n = lane & 15;
  const int mb = (lane >> 4) << 3;
  float rs[8];
#pragma unroll
  for (int r = 0; r < 8; ++r) rs[r] = rsum[row0 + mb + r];
#pragma unroll
  for (int cb = 0; cb < 8; ++cb)
#pragma unroll
    for (int r = 0; r < 8; ++r) {
      float v = acc[cb][r] / rs[r];
      v = v > 0.f ? v : (__expf(v) - 1.f);   // elu
      long idx = (long)(row0 + mb + r) * HD + (cb << 4) + n;
      out[idx] = v + h_w[idx];
    }
}

// ---------------- launch ----------------

extern "C" void kernel_launch(void* const* d_in, const int* in_sizes, int n_in,
                              void* d_out, int out_size, void* d_ws, size_t ws_size,
                              hipStream_t stream) {
  const float* Xw      = (const float*)d_in[0];
  const float* Xt      = (const float*)d_in[1];
  const float* ww_adj  = (const float*)d_in[2];
  const float* wt_adj  = (const float*)d_in[3];
  const float* Wp_w    = (const float*)d_in[4];
  const float* bp_w    = (const float*)d_in[5];
  const float* Wp_t    = (const float*)d_in[6];
  const float* bp_t    = (const float*)d_in[7];
  const float* W_worker= (const float*)d_in[8];
  const float* b_worker= (const float*)d_in[9];
  const float* W_task  = (const float*)d_in[10];
  const float* b_task  = (const float*)d_in[11];
  const float* W_ww    = (const float*)d_in[12];
  const float* W_wt    = (const float*)d_in[13];
  const float* W_tw    = (const float*)d_in[14];
  const float* W_igat  = (const float*)d_in[15];
  const float* a_src   = (const float*)d_in[16];
  const float* a_dst   = (const float*)d_in[17];

  float* out_worker = (float*)d_out;
  float* out_task   = (float*)d_out + (long)NW * HD;

  size_t off = 0;
  char* ws = (char*)d_ws;
  auto carve = [&](size_t bytes) -> void* {
    void* p = ws + off;
    off += (bytes + 255) & ~(size_t)255;
    return p;
  };
  float*  h_w   = (float*)carve((size_t)NW * HD * 4);
  float*  h_t   = (float*)carve((size_t)NT * HD * 4);
  float*  s_w   = (float*)carve((size_t)NW * HD * 4);
  float*  s_t   = (float*)carve((size_t)NT * HD * 4);
  float*  rgcnF = (float*)carve((size_t)NW * HD * 4);
  float*  Wh    = (float*)carve((size_t)NW * HD * 4);
  __bf16* m_ww  = (__bf16*)carve((size_t)NW * HD * 2);
  __bf16* m_wt  = (__bf16*)carve((size_t)NW * HD * 2);
  __bf16* m_tw  = (__bf16*)carve((size_t)NT * HD * 2);
  __bf16* WhB   = (__bf16*)carve((size_t)NW * HD * 2);
  __bf16* wtT   = (__bf16*)carve((size_t)NT * NW * 2);   // wt_adj^T, packed-A bf16
  float*  f_src = (float*)carve((size_t)NW * 4);
  float*  f_dst = (float*)carve((size_t)NW * 4);
  float*  rmax  = (float*)carve((size_t)NW * 4);
  float*  rsum  = (float*)carve((size_t)NW * 4);
  // packed bf16 weights (128x128 each)
  __bf16* pWp_w    = (__bf16*)carve((size_t)HD * HD * 2);
  __bf16* pWp_t    = (__bf16*)carve((size_t)HD * HD * 2);
  __bf16* pW_worker= (__bf16*)carve((size_t)HD * HD * 2);
  __bf16* pW_task  = (__bf16*)carve((size_t)HD * HD * 2);
  __bf16* pW_ww    = (__bf16*)carve((size_t)HD * HD * 2);
  __bf16* pW_wt    = (__bf16*)carve((size_t)HD * HD * 2);
  __bf16* pW_tw    = (__bf16*)carve((size_t)HD * HD * 2);
  __bf16* pW_igat  = (__bf16*)carve((size_t)HD * HD * 2);
  (void)ws_size; (void)in_sizes; (void)n_in; (void)out_size;

  const dim3 blk(32, 8);                 // 8 wave32 per block, 1 tile (or strip) per wave
  const dim3 gw(HD / 16, NW / 128);      // tile grids
  const dim3 gt(HD / 16, NT / 128);
  const dim3 pk(HD * HD / 256);          // weight pack grid (64 blocks)

  // Stage -1: pack all B-side weights; transpose-pack wt_adj^T for the task RGCN
  pack_b_kernel<<<pk, 256, 0, stream>>>(Wp_w,     pWp_w);
  pack_b_kernel<<<pk, 256, 0, stream>>>(Wp_t,     pWp_t);
  pack_b_kernel<<<pk, 256, 0, stream>>>(W_worker, pW_worker);
  pack_b_kernel<<<pk, 256, 0, stream>>>(W_task,   pW_task);
  pack_b_kernel<<<pk, 256, 0, stream>>>(W_ww,     pW_ww);
  pack_b_kernel<<<pk, 256, 0, stream>>>(W_wt,     pW_wt);
  pack_b_kernel<<<pk, 256, 0, stream>>>(W_tw,     pW_tw);
  pack_b_kernel<<<pk, 256, 0, stream>>>(W_igat,   pW_igat);
  transpose_pack_kernel<<<dim3((unsigned)(((long)NW * NT) / 256)), 256, 0, stream>>>(wt_adj, wtT);

  // Stage 0: input projections
  proj_relu_kernel<<<gw, blk, 0, stream>>>(Xw, pWp_w, bp_w, h_w);
  proj_relu_kernel<<<gt, blk, 0, stream>>>(Xt, pWp_t, bp_t, h_t);

  // Stage 1: per-type 128x128 transforms
  gemm128_kernel<<<gw, blk, 0, stream>>>(h_w, pW_ww,     nullptr,  nullptr, m_ww);
  gemm128_kernel<<<gw, blk, 0, stream>>>(h_w, pW_wt,     nullptr,  nullptr, m_wt);
  gemm128_kernel<<<gt, blk, 0, stream>>>(h_t, pW_tw,     nullptr,  nullptr, m_tw);
  gemm128_kernel<<<gw, blk, 0, stream>>>(h_w, pW_worker, b_worker, s_w,     nullptr);
  gemm128_kernel<<<gt, blk, 0, stream>>>(h_t, pW_task,   b_task,   s_t,     nullptr);

  // Stage 2: adjacency contractions, 16x128 strips (adjacency streamed once)
  rgcn_worker_kernel<<<dim3(NW / 128), blk, 0, stream>>>(s_w, ww_adj, m_ww, wt_adj, m_tw, rgcnF);
  rgcn_task_kernel<<<dim3(NT / 128), blk, 0, stream>>>(s_t, wtT, m_wt, h_t, out_task);

  // Stage 3: Wh = h_w_rgcn @ W_igat (f32 for f-vectors, packed bf16 for attention)
  gemm128_kernel<<<gw, blk, 0, stream>>>(rgcnF, pW_igat, nullptr, Wh, WhB);

  // Stage 4: attention features + masked-softmax row statistics
  fvec_kernel<<<dim3(NW / 256), dim3(256), 0, stream>>>(Wh, a_src, a_dst, f_src, f_dst, NW);
  softmax_stats_kernel<<<dim3(NW), dim3(256), 0, stream>>>(ww_adj, f_src, f_dst, rmax, rsum);

  // Stage 5: fused attention @ Wh with ELU + residual -> final_worker
  attn_kernel<<<dim3(NW / 128), blk, 0, stream>>>(ww_adj, WhB, f_src, f_dst, rmax, rsum, h_w, out_worker);
}
